// NeighSim_50878182588702
// MI455X (gfx1250) — matrix operand order repaired and checked by reading, not compiled, not run on hardware
//
#include <hip/hip_runtime.h>

typedef __attribute__((ext_vector_type(16))) _Float16 v16h;
typedef __attribute__((ext_vector_type(8)))  float    v8f;

#define EPSF   1e-8f
#define HW     256
#define NC     32
#define NB     8
#define TILE_W 16
#define TILE_H 8
#define HALO_W 18
#define HALO_H 10
#define NPIX   (HALO_W * HALO_H)        // 180 halo pixels
#define FSTR   40                       // f16 channel stride per pixel (80B, 16B aligned)
#define NSTAGE (HALO_H * NC * HALO_W)   // 5760 staged f32 elements

// 8 neighbor directions, matching _directions(): y outer, x inner, center removed
__device__ __constant__ int c_DX[8] = {-1, 0, 1, -1, 1, -1, 0, 1};
__device__ __constant__ int c_DY[8] = {-1, -1, -1, 0, 0, 1, 1, 1};

// 8-way per-lane mux over a WMMA f32 accumulator fragment (index k = lane&7)
__device__ __forceinline__ float laneSel8(v8f c, int k) {
    float s0 = (k & 1) ? c[1] : c[0];
    float s1 = (k & 1) ? c[3] : c[2];
    float s2 = (k & 1) ? c[5] : c[4];
    float s3 = (k & 1) ? c[7] : c[6];
    float t0 = (k & 2) ? s1 : s0;
    float t1 = (k & 2) ? s3 : s2;
    return (k & 4) ? t1 : t0;
}

// One relaxation step: async global->LDS staging, cosine sims via WMMA f16
// (K=32=C), mask-normalized mean sim map, and (if do_update) the
// softmax-weighted neighbor average into fout.
__global__ __launch_bounds__(256)
void neighsim_iter(const float* __restrict__ fin, float* __restrict__ fout,
                   float* __restrict__ simmap, int do_update)
{
    __shared__ float    sStage[NSTAGE];             // async-staged f32 halo, [r][c][xx]
    __shared__ _Float16 sF[NPIX * FSTR];            // f16 halo tile, [pixel][channel]
    __shared__ float    sN[NPIX];                   // per-pixel L2 norm
    __shared__ float    sW[TILE_W * TILE_H * 8];    // softmax weights per tile pixel

    const int tid = threadIdx.x;
    const int b  = blockIdx.z;
    const int x0 = blockIdx.x * TILE_W;
    const int y0 = blockIdx.y * TILE_H;

    // ---- Async stage: global f32 halo -> LDS (bypasses VGPRs; ASYNCcnt) ----
    // Out-of-image lanes are EXEC-masked off; their slots are zeroed in the
    // convert pass below, giving the reference's zero-padded shift semantics.
    const uint32_t ldsBase = (uint32_t)(size_t)(&sStage[0]);
    for (int i = tid; i < NSTAGE; i += 256) {
        int r   = i / (NC * HALO_W);
        int rem = i - r * (NC * HALO_W);
        int c   = rem / HALO_W;
        int xx  = rem - c * HALO_W;
        int gx = x0 - 1 + xx, gy = y0 - 1 + r;
        if (gx >= 0 && gx < HW && gy >= 0 && gy < HW) {
            const float* gp = &fin[(((size_t)b * NC + c) * HW + gy) * HW + gx];
            uint64_t ga = (uint64_t)(size_t)gp;
            uint32_t la = ldsBase + (uint32_t)i * 4u;
            asm volatile("global_load_async_to_lds_b32 %0, %1, off"
                         :: "v"(la), "v"(ga) : "memory");
        }
    }
    asm volatile("s_wait_asynccnt 0" ::: "memory");  // drain before barrier
    __syncthreads();

    // ---- Convert staged f32 -> f16 tile (zero outside image) ----
    for (int i = tid; i < NSTAGE; i += 256) {
        int r   = i / (NC * HALO_W);
        int rem = i - r * (NC * HALO_W);
        int c   = rem / HALO_W;
        int xx  = rem - c * HALO_W;
        int gx = x0 - 1 + xx, gy = y0 - 1 + r;
        float v = (gx >= 0 && gx < HW && gy >= 0 && gy < HW) ? sStage[i] : 0.0f;
        sF[(r * HALO_W + xx) * FSTR + c] = (_Float16)v;
    }
    __syncthreads();

    // ---- Per-pixel norms from the f16 tile ----
    for (int i = tid; i < NPIX; i += 256) {
        float s = 0.0f;
#pragma unroll
        for (int c = 0; c < NC; ++c) { float v = (float)sF[i * FSTR + c]; s += v * v; }
        sN[i] = sqrtf(s);
    }
    __syncthreads();

    // ---- Wave w owns row y0+w; 16 pixels per wave, fragments per ISA layout ----
    const int wv   = tid >> 5;
    const int lane = tid & 31;
    const int j    = lane & 15;                       // pixel column (== M for A, N for B)
    const int hpC  = (wv + 1) * HALO_W + (j + 1);     // center halo pixel for this M/N

    // A (16x32 f16): lanes 0-15 carry K {0..7,16..23}; lanes 16-31 K {8..15,24..31}
    const int k0 = (lane < 16) ? 0 : 8;
    v16h a;
#pragma unroll
    for (int e = 0; e < 8; ++e) a[e]     = sF[hpC * FSTR + k0 + e];
#pragma unroll
    for (int e = 0; e < 8; ++e) a[8 + e] = sF[hpC * FSTR + k0 + 16 + e];

    const int cb = k0;                                // B: same K order as A
    const float np = sN[hpC];

    float sims[8];
    float tot = 0.0f;
#pragma unroll
    for (int d = 0; d < 8; ++d) {
        const int dx = c_DX[d], dy = c_DY[d];
        const int hq = (wv + 1 + dy) * HALO_W + (j + 1 + dx);

        // B (32x16 f16): lane holds column N=j, K {cb..cb+7, cb+16..cb+23}
        v16h bf;
#pragma unroll
        for (int e = 0; e < 8; ++e) bf[e] = sF[hq * FSTR + cb + e];
#pragma unroll
        for (int e = 0; e < 8; ++e) bf[8 + e] = sF[hq * FSTR + cb + 16 + e];

        v8f acc = {};
        acc = __builtin_amdgcn_wmma_f32_16x16x32_f16(
                  false, a, false, bf, (short)0, acc, false, false);

        // Diagonal (q,q): lane q (q<8) / lane q+16 (q>=8), element q%8.
        float owned = laneSel8(acc, lane & 7);
        int   src   = (lane < 8) ? lane : (lane + 16);
        float dot   = __int_as_float(
            __builtin_amdgcn_ds_bpermute(src << 2, __float_as_int(owned)));

        // Masked cosine sim: zero-padded halo gives dot=0, nq=0 -> sim=0 at borders
        float nq = sN[hq];
        float s  = dot / fmaxf(np * nq, EPSF);
        sims[d] = s;
        tot += s;
    }

    // ---- total_sim = sum(sim*m) / sum(m); cnt = vx*vy - 1 ----
    const int gx = x0 + j, gy = y0 + wv;
    if (lane < 16) {
        int vx = 1 + (gx > 0) + (gx < HW - 1);
        int vy = 1 + (gy > 0) + (gy < HW - 1);
        simmap[((size_t)b * HW + gy) * HW + gx] = tot / (float)(vx * vy - 1);
    }

    // ---- Softmax weights + feature update ----
    if (do_update) {
        float mx = sims[0];
#pragma unroll
        for (int d = 1; d < 8; ++d) mx = fmaxf(mx, sims[d]);
        float e[8], es = 0.0f;
#pragma unroll
        for (int d = 0; d < 8; ++d) { e[d] = expf(sims[d] - mx); es += e[d]; }
        float inv = 1.0f / es;
        if (lane < 16) {
#pragma unroll
            for (int d = 0; d < 8; ++d) sW[(wv * 16 + j) * 8 + d] = e[d] * inv;
        }
    }
    __syncthreads();
    if (do_update) {
        // lane -> (pixel j, channel c): 16 px x 2 ch per step, 16 steps covers C=32
#pragma unroll
        for (int cc = 0; cc < 16; ++cc) {
            int c = cc * 2 + (lane >> 4);
            float accu = 0.0f;
#pragma unroll
            for (int d = 0; d < 8; ++d) {
                int hq = (wv + 1 + c_DY[d]) * HALO_W + (j + 1 + c_DX[d]);
                accu += sW[(wv * 16 + j) * 8 + d] * (float)sF[hq * FSTR + c];
            }
            fout[(((size_t)b * NC + c) * HW + gy) * HW + gx] = accu;
        }
    }
}

// out[g,t,y,x] = mean over 4 batches of map{0,0,1,2}[b,y,x]
__global__ __launch_bounds__(256)
void neighsim_final(const float* __restrict__ maps, float* __restrict__ out, int n)
{
    int i = blockIdx.x * blockDim.x + threadIdx.x;
    if (i >= n) return;
    int pix = i & 0xFFFF;
    int t   = (i >> 16) & 3;
    int g   = i >> 18;
    int m   = (t <= 1) ? 0 : (t - 1);
    const float* mp = maps + (size_t)m * NB * HW * HW;
    float s = 0.0f;
#pragma unroll
    for (int k = 0; k < 4; ++k)
        s += mp[(size_t)(g * 4 + k) * (HW * HW) + pix];
    out[i] = 0.25f * s;
}

extern "C" void kernel_launch(void* const* d_in, const int* in_sizes, int n_in,
                              void* d_out, int out_size, void* d_ws, size_t ws_size,
                              hipStream_t stream) {
    (void)in_sizes; (void)n_in; (void)ws_size;
    const float* f0 = (const float*)d_in[0];
    char* ws = (char*)d_ws;

    const size_t featBytes = (size_t)NB * NC * HW * HW * sizeof(float);  // 64 MB
    const size_t mapElems  = (size_t)NB * HW * HW;                       // per-map floats
    float* f1   = (float*)ws;
    float* f2   = (float*)(ws + featBytes);
    float* maps = (float*)(ws + 2 * featBytes);   // 3 maps, 6 MB (ws >= ~134 MB)

    dim3 grid(HW / TILE_W, HW / TILE_H, NB);
    dim3 blk(256);
    // maps: ts(f0), ts(f1), ts(f2); third step needs no feature update
    neighsim_iter<<<grid, blk, 0, stream>>>(f0, f1, maps,                1);
    neighsim_iter<<<grid, blk, 0, stream>>>(f1, f2, maps + mapElems,     1);
    neighsim_iter<<<grid, blk, 0, stream>>>(f2, nullptr, maps + 2 * mapElems, 0);

    neighsim_final<<<(out_size + 255) / 256, 256, 0, stream>>>(maps, (float*)d_out, out_size);
}
